// MultiHeadedAttention_24249385353218
// MI455X (gfx1250) — matrix operand order repaired and checked by reading, not compiled, not run on hardware
//
#include <hip/hip_runtime.h>
#include <hip/hip_bf16.h>
#include <stdint.h>

#define NHEAD   16
#define DK      64
#define DMODEL  1024
#define BATCH   4
#define SEQ     2048
#define MTOT    (BATCH * SEQ)

typedef __attribute__((ext_vector_type(16))) __bf16 v16bf;
typedef __attribute__((ext_vector_type(8)))  float  v8f;
typedef __attribute__((ext_vector_type(4)))  int    v4i;

union FragA {                 // 16x32 bf16 A/B operand = 8 VGPRs
    v16bf v;
    uint4 q[2];
};

// Opportunistic async global->LDS (gfx1250 GLOBAL_LOAD_ASYNC_TO_LDS_B128).
#if defined(__has_builtin)
#if __has_builtin(__builtin_amdgcn_global_load_async_to_lds_b128) && \
    __has_builtin(__builtin_amdgcn_s_wait_asynccnt)
#define HAVE_ASYNC_LDS 1
#endif
#endif
#ifndef HAVE_ASYNC_LDS
#define HAVE_ASYNC_LDS 0
#endif

typedef __attribute__((address_space(1))) v4i g1_v4i;
typedef __attribute__((address_space(3))) v4i l3_v4i;

// round-to-nearest-even fp32 -> bf16, packed pair
__device__ __forceinline__ uint32_t pk_bf16(float lo, float hi) {
    union { float f; uint32_t u; } a, b;
    a.f = lo; b.f = hi;
    uint32_t ra = a.u + 0x7FFFu + ((a.u >> 16) & 1u);
    uint32_t rb = b.u + 0x7FFFu + ((b.u >> 16) & 1u);
    return (ra >> 16) | (rb & 0xFFFF0000u);
}
__device__ __forceinline__ uint16_t to_bf16(float x) {
    return (uint16_t)(pk_bf16(x, 0.0f) & 0xFFFFu);
}

// ---------------------------------------------------------------------------
// GEMM: Y[m][n] = sum_k A[m][k] * W[n][k] + bias[n]
// MODE 0: A fp32, Y bf16 -> [B,H,S,DK]          (Q, K projections)
// MODE 1: A fp32, Y bf16 -> [B,H,DK,S]          (V projection, transposed)
// MODE 2: A bf16, Y fp32 -> [m][n] plain        (output projection)
// Block: 256 threads / 8 waves, tile 128(M) x 128(N); wave = 32x64
// (2x4 WMMA accumulators, 8 WMMA / K-step). K step 32, software-pipelined.
// ---------------------------------------------------------------------------
template <int MODE>
__global__ __launch_bounds__(256)
void gemm_kernel(const void* __restrict__ Xv, const float* __restrict__ W,
                 const float* __restrict__ bias, void* __restrict__ Yv)
{
    constexpr int LDA = 48;                       // 32 data + 16 pad bf16 (96B rows)
    __shared__ uint16_t lds_a[128 * LDA];
    __shared__ uint16_t lds_b[128 * LDA];

    const int m0   = blockIdx.x * 128;
    const int n0   = blockIdx.y * 128;
    const int tid  = threadIdx.x;
    const int lane = tid & 31;
    const int wave = tid >> 5;
    const int wm   = (wave >> 1) * 32;            // 4(M) x 2(N) wave grid
    const int wn   = (wave & 1) * 64;

    const float*    Xf = (const float*)Xv;
    const uint16_t* Xh = (const uint16_t*)Xv;

    // staging: each thread owns one 16-element row segment of A and of B
    const int ar = tid >> 1;                      // row 0..127
    const int ac = (tid & 1) * 16;                // col segment 0 / 16

    const int lrow  = lane & 15;
    const int lhalf = lane >> 4;

    v8f acc[2][4] = {};
    float4 fA[4], fB[4];
    uint4  uA[2];

    auto g_load = [&](int kk) {
        if (MODE == 2) {
#if !HAVE_ASYNC_LDS
            const uint4* s = (const uint4*)(Xh + (size_t)(m0 + ar) * DMODEL + kk + ac);
            uA[0] = s[0]; uA[1] = s[1];
#endif
        } else {
            const float4* s = (const float4*)(Xf + (size_t)(m0 + ar) * DMODEL + kk + ac);
            fA[0] = s[0]; fA[1] = s[1]; fA[2] = s[2]; fA[3] = s[3];
        }
        const float4* sb = (const float4*)(W + (size_t)(n0 + ar) * DMODEL + kk + ac);
        fB[0] = sb[0]; fB[1] = sb[1]; fB[2] = sb[2]; fB[3] = sb[3];
        if (kk + 64 < DMODEL) {                   // stream next-next tile into L2
            if (MODE != 2)
                __builtin_prefetch(Xf + (size_t)(m0 + ar) * DMODEL + kk + 64 + ac, 0, 0);
            __builtin_prefetch(W + (size_t)(n0 + ar) * DMODEL + kk + 64 + ac, 0, 0);
        }
    };

    auto lds_store = [&](int kk) {
        if (MODE == 2) {
#if HAVE_ASYNC_LDS
            // A is already bf16: pure DMA copy, no VALU conversion needed.
            g1_v4i* gsrc = (g1_v4i*)(Xh + (size_t)(m0 + ar) * DMODEL + kk + ac);
            l3_v4i* ldst = (l3_v4i*)(lds_a + ar * LDA + ac);
            __builtin_amdgcn_global_load_async_to_lds_b128(gsrc,     ldst,     0, 0);
            __builtin_amdgcn_global_load_async_to_lds_b128(gsrc + 1, ldst + 1, 0, 0);
            __builtin_amdgcn_s_wait_asynccnt(0);
#else
            *(uint4*)(lds_a + ar * LDA + ac)     = uA[0];
            *(uint4*)(lds_a + ar * LDA + ac + 8) = uA[1];
#endif
        } else {
            uint4 u0, u1;
            u0.x = pk_bf16(fA[0].x, fA[0].y); u0.y = pk_bf16(fA[0].z, fA[0].w);
            u0.z = pk_bf16(fA[1].x, fA[1].y); u0.w = pk_bf16(fA[1].z, fA[1].w);
            u1.x = pk_bf16(fA[2].x, fA[2].y); u1.y = pk_bf16(fA[2].z, fA[2].w);
            u1.z = pk_bf16(fA[3].x, fA[3].y); u1.w = pk_bf16(fA[3].z, fA[3].w);
            *(uint4*)(lds_a + ar * LDA + ac)     = u0;
            *(uint4*)(lds_a + ar * LDA + ac + 8) = u1;
        }
        uint4 b0, b1;
        b0.x = pk_bf16(fB[0].x, fB[0].y); b0.y = pk_bf16(fB[0].z, fB[0].w);
        b0.z = pk_bf16(fB[1].x, fB[1].y); b0.w = pk_bf16(fB[1].z, fB[1].w);
        b1.x = pk_bf16(fB[2].x, fB[2].y); b1.y = pk_bf16(fB[2].z, fB[2].w);
        b1.z = pk_bf16(fB[3].x, fB[3].y); b1.w = pk_bf16(fB[3].z, fB[3].w);
        *(uint4*)(lds_b + ar * LDA + ac)     = b0;
        *(uint4*)(lds_b + ar * LDA + ac + 8) = b1;
    };

    g_load(0);
    for (int kk = 0; kk < DMODEL; kk += 32) {
        __syncthreads();                          // previous compute done with LDS
        lds_store(kk);
        __syncthreads();                          // tile visible to all waves

        if (kk + 32 < DMODEL) g_load(kk + 32);    // overlap next load with WMMAs

        FragA a[2], b[4];
#pragma unroll
        for (int i = 0; i < 2; ++i) {             // A 16x32: K 0-7/16-23 | 8-15/24-31
            const uint16_t* p = lds_a + (wm + i * 16 + lrow) * LDA;
            a[i].q[0] = *(const uint4*)(p + lhalf * 8);
            a[i].q[1] = *(const uint4*)(p + 16 + lhalf * 8);
        }
#pragma unroll
        for (int j = 0; j < 4; ++j) {             // B 32x16: K 0-7/16-23 | 8-15/24-31
            const uint16_t* p = lds_b + (wn + j * 16 + lrow) * LDA;
            b[j].q[0] = *(const uint4*)(p + lhalf * 8);
            b[j].q[1] = *(const uint4*)(p + 16 + lhalf * 8);
        }
#pragma unroll
        for (int i = 0; i < 2; ++i)
#pragma unroll
            for (int j = 0; j < 4; ++j)
                acc[i][j] = __builtin_amdgcn_wmma_f32_16x16x32_bf16(
                    false, a[i].v, false, b[j].v, (short)0, acc[i][j], false, false);
    }

    // ---- epilogue: bias + layout-specific store (all index math hoisted) ----
    const int moff   = (lane >> 4) * 8;           // C layout: M = r + 8*(lane>=16)
    const int bb     = m0 / SEQ;                  // 128 | 2048: tile in one batch
    const int s_base = (m0 - bb * SEQ) + wm + moff;
    const int h      = (n0 + wn) >> 6;            // 64-wide wave tile: one head
#pragma unroll
    for (int j = 0; j < 4; ++j) {
        const int n  = n0 + wn + j * 16 + lrow;
        const int d  = j * 16 + lrow;
        const float bvv = bias[n];
#pragma unroll
        for (int i = 0; i < 2; ++i) {
#pragma unroll
            for (int r = 0; r < 8; ++r) {
                const float val = acc[i][j][r] + bvv;
                const int s = s_base + i * 16 + r;
                if (MODE == 2) {
                    ((float*)Yv)[(size_t)(m0 + wm + moff + i * 16 + r) * DMODEL + n] = val;
                } else if (MODE == 0) {
                    ((uint16_t*)Yv)[(((size_t)(bb * NHEAD + h)) * SEQ + s) * DK + d]
                        = to_bf16(val);
                } else {
                    ((uint16_t*)Yv)[(((size_t)(bb * NHEAD + h)) * DK + d) * SEQ + s]
                        = to_bf16(val);
                }
            }
        }
    }
}

// ---------------------------------------------------------------------------
// Flash attention: one wave = 16 query rows, streams 64 keys/iteration
// (16 WMMA per chunk). Softmax runs in the exp2 domain so the hardware exp2
// is used directly. Q,K: [B*H,S,64] bf16; VT: [B*H,64,S] bf16; X: [B,S,H*64].
// ---------------------------------------------------------------------------
__global__ __launch_bounds__(256)
void attn_kernel(const uint16_t* __restrict__ Q, const uint16_t* __restrict__ K,
                 const uint16_t* __restrict__ VT, uint16_t* __restrict__ X)
{
    constexpr int LDP = 72;                       // 64 data + 8 pad (144B rows)
    __shared__ uint16_t p_lds[8][16 * LDP];       // per-wave P tile (C->A relayout)

    const int tid   = threadIdx.x;
    const int lane  = tid & 31;
    const int wave  = tid >> 5;
    const int bh    = blockIdx.x >> 4;
    const int qs0   = (blockIdx.x & 15) * 128 + wave * 16;
    const int lrow  = lane & 15;
    const int lhalf = lane >> 4;

    const uint16_t* Kb  = K  + (size_t)bh * SEQ * DK;
    const uint16_t* VTb = VT + (size_t)bh * DK * SEQ;

    // Q A-fragments for d windows [0,32) and [32,64)
    FragA qa[2];
#pragma unroll
    for (int w = 0; w < 2; ++w) {
        const uint16_t* p = Q + ((size_t)bh * SEQ + qs0 + lrow) * DK + w * 32;
        qa[w].q[0] = *(const uint4*)(p + lhalf * 8);
        qa[w].q[1] = *(const uint4*)(p + 16 + lhalf * 8);
    }

    v8f   o[4] = {};
    float mrun[8], lrun[8];
#pragma unroll
    for (int r = 0; r < 8; ++r) { mrun[r] = -1e30f; lrun[r] = 0.0f; }
    const float scale2 = 0.125f * 1.44269504f;    // (1/sqrt(dk)) * log2(e)

    for (int kc = 0; kc < SEQ; kc += 64) {
        if (kc + 64 < SEQ)
            __builtin_prefetch(Kb + (size_t)(kc + 64 + lrow) * DK, 0, 0);

        // ---- scores S = Q K^T (16 x 64) : 8 WMMA ----
        v8f s4[4] = {};
#pragma unroll
        for (int j = 0; j < 4; ++j)
#pragma unroll
            for (int w = 0; w < 2; ++w) {
                FragA kb;
                const uint16_t* p = Kb + (size_t)(kc + j * 16 + lrow) * DK
                                       + w * 32;
                kb.q[0] = *(const uint4*)(p + lhalf * 8);
                kb.q[1] = *(const uint4*)(p + 16 + lhalf * 8);
                s4[j] = __builtin_amdgcn_wmma_f32_16x16x32_bf16(
                            false, qa[w].v, false, kb.v, (short)0, s4[j], false, false);
            }

        // ---- online softmax over the 64-key chunk (per accumulator row) ----
#pragma unroll
        for (int r = 0; r < 8; ++r) {
            const float x0 = s4[0][r] * scale2, x1 = s4[1][r] * scale2;
            const float x2 = s4[2][r] * scale2, x3 = s4[3][r] * scale2;
            float t = fmaxf(fmaxf(x0, x1), fmaxf(x2, x3));
#pragma unroll
            for (int m = 8; m >= 1; m >>= 1)
                t = fmaxf(t, __shfl_xor(t, m, 32));     // 16-lane group reduce
            const float mnew = fmaxf(mrun[r], t);
            const float corr = exp2f(mrun[r] - mnew);
            const float p0 = exp2f(x0 - mnew);
            const float p1 = exp2f(x1 - mnew);
            const float p2 = exp2f(x2 - mnew);
            const float p3 = exp2f(x3 - mnew);
            float psum = (p0 + p1) + (p2 + p3);
#pragma unroll
            for (int m = 8; m >= 1; m >>= 1)
                psum += __shfl_xor(psum, m, 32);
            lrun[r] = lrun[r] * corr + psum;
            mrun[r] = mnew;
            o[0][r] *= corr; o[1][r] *= corr; o[2][r] *= corr; o[3][r] *= corr;
            // scatter P into wave-private LDS (row = r + 8*lhalf, col = key)
            uint16_t* pl = &p_lds[wave][(r + lhalf * 8) * LDP];
            pl[lrow]      = to_bf16(p0);
            pl[lrow + 16] = to_bf16(p1);
            pl[lrow + 32] = to_bf16(p2);
            pl[lrow + 48] = to_bf16(p3);
        }
        asm volatile("s_wait_dscnt 0" ::: "memory");    // same-wave LDS RAW

        // P as two A-fragments (16 x 64)
        FragA pa[2];
#pragma unroll
        for (int w = 0; w < 2; ++w) {
            const uint16_t* p = &p_lds[wave][lrow * LDP + w * 32];
            pa[w].q[0] = *(const uint4*)(p + lhalf * 8);
            pa[w].q[1] = *(const uint4*)(p + 16 + lhalf * 8);
        }

        // ---- O += P V (16 x 64) : 8 WMMA, V read transposed/contiguous ----
#pragma unroll
        for (int j = 0; j < 4; ++j)
#pragma unroll
            for (int w = 0; w < 2; ++w) {
                FragA vb;
                const uint16_t* p = VTb + (size_t)(j * 16 + lrow) * SEQ
                                        + kc + w * 32;
                vb.q[0] = *(const uint4*)(p + lhalf * 8);
                vb.q[1] = *(const uint4*)(p + 16 + lhalf * 8);
                o[j] = __builtin_amdgcn_wmma_f32_16x16x32_bf16(
                           false, pa[w].v, false, vb.v, (short)0, o[j], false, false);
            }
    }

    // ---- normalize and store to [B, S, H*64] bf16 ----
    const int b = bh / NHEAD, h = bh % NHEAD;
    uint16_t* Xb = X + ((size_t)b * SEQ + qs0 + lhalf * 8) * DMODEL + h * DK;
#pragma unroll
    for (int r = 0; r < 8; ++r) {
        const float inv = 1.0f / lrun[r];
#pragma unroll
        for (int j = 0; j < 4; ++j)
            Xb[(size_t)r * DMODEL + j * 16 + lrow] = to_bf16(o[j][r] * inv);
    }
}

// ---------------------------------------------------------------------------
extern "C" void kernel_launch(void* const* d_in, const int* in_sizes, int n_in,
                              void* d_out, int out_size, void* d_ws, size_t ws_size,
                              hipStream_t stream)
{
    const float* query = (const float*)d_in[0];
    const float* key   = (const float*)d_in[1];
    const float* value = (const float*)d_in[2];
    const float* Wq    = (const float*)d_in[3];
    const float* bq    = (const float*)d_in[4];
    const float* Wk    = (const float*)d_in[5];
    const float* bk    = (const float*)d_in[6];
    const float* Wv    = (const float*)d_in[7];
    const float* bv    = (const float*)d_in[8];
    const float* Wo    = (const float*)d_in[9];
    const float* bo    = (const float*)d_in[10];

    const size_t elems = (size_t)MTOT * DMODEL;
    uint16_t* qbuf  = (uint16_t*)d_ws;            // [B,H,S,64] bf16
    uint16_t* kbuf  = qbuf  + elems;              // [B,H,S,64] bf16
    uint16_t* vtbuf = kbuf  + elems;              // [B,H,64,S] bf16
    uint16_t* xbuf  = vtbuf + elems;              // [B,S,1024] bf16

    dim3 gg(MTOT / 128, DMODEL / 128);
    gemm_kernel<0><<<gg, 256, 0, stream>>>(query, Wq, bq, qbuf);
    gemm_kernel<0><<<gg, 256, 0, stream>>>(key,   Wk, bk, kbuf);
    gemm_kernel<1><<<gg, 256, 0, stream>>>(value, Wv, bv, vtbuf);

    attn_kernel<<<dim3(BATCH * NHEAD * (SEQ / 128)), 256, 0, stream>>>(
        qbuf, kbuf, vtbuf, xbuf);

    gemm_kernel<2><<<gg, 256, 0, stream>>>(xbuf, Wo, bo, d_out);
}
